// AutoRegressiveInferenceNet_12051678232919
// MI455X (gfx1250) — matrix operand order repaired and checked by reading, not compiled, not run on hardware
//
#include <hip/hip_runtime.h>
#include <cstdint>
#include <cstddef>

#define BATCH 16384
#define DIM   1024    // IN == D == 1024

typedef __bf16 bf16_t;
typedef __attribute__((ext_vector_type(16))) __bf16 v16bf;
typedef __attribute__((ext_vector_type(8)))  float  v8f;

// ---------------------------------------------------------------------------
// Build a WMMA A-fragment (16x32 bf16, contiguous per-lane k order, matching B)
// from an fp32 row, applying value -> 2*value - 1 on the fly.
// base_k = k0 + (lane>=16 ? 16 : 0); elements e0..15 -> K base_k..base_k+15
// (contiguous).  All addresses 32B aligned.
// ---------------------------------------------------------------------------
static __device__ __forceinline__ v16bf load_a_2xm1(const float* __restrict__ rowp,
                                                    int base_k) {
  const float4* p = (const float4*)(rowp + base_k);
  float4 f0 = p[0], f1 = p[1];
  const float4* q = (const float4*)(rowp + base_k + 8);
  float4 f2 = q[0], f3 = q[1];
  v16bf a;
  a[0]  = (__bf16)(2.0f * f0.x - 1.0f);
  a[1]  = (__bf16)(2.0f * f0.y - 1.0f);
  a[2]  = (__bf16)(2.0f * f0.z - 1.0f);
  a[3]  = (__bf16)(2.0f * f0.w - 1.0f);
  a[4]  = (__bf16)(2.0f * f1.x - 1.0f);
  a[5]  = (__bf16)(2.0f * f1.y - 1.0f);
  a[6]  = (__bf16)(2.0f * f1.z - 1.0f);
  a[7]  = (__bf16)(2.0f * f1.w - 1.0f);
  a[8]  = (__bf16)(2.0f * f2.x - 1.0f);
  a[9]  = (__bf16)(2.0f * f2.y - 1.0f);
  a[10] = (__bf16)(2.0f * f2.z - 1.0f);
  a[11] = (__bf16)(2.0f * f2.w - 1.0f);
  a[12] = (__bf16)(2.0f * f3.x - 1.0f);
  a[13] = (__bf16)(2.0f * f3.y - 1.0f);
  a[14] = (__bf16)(2.0f * f3.z - 1.0f);
  a[15] = (__bf16)(2.0f * f3.w - 1.0f);
  return a;
}

// ---------------------------------------------------------------------------
// Kernel P: convert W0 and W1 (both [1024][1024] fp32, row-major) to bf16.
// Row-major bf16 weights serve directly as WMMA B-fragments:
//   B[k][n] = W[n][k] -> per lane (col n = lane&15) K is contiguous in memory.
// ---------------------------------------------------------------------------
__global__ void __launch_bounds__(256)
prep_bf16(const float* __restrict__ W0, const float* __restrict__ W1,
          bf16_t* __restrict__ W0b, bf16_t* __restrict__ W1b) {
  size_t i = (size_t)blockIdx.x * 256 + threadIdx.x;
  if (i < (size_t)DIM * DIM) {
    W0b[i] = (__bf16)W0[i];
    W1b[i] = (__bf16)W1[i];
  }
}

// ---------------------------------------------------------------------------
// Generic bf16 WMMA GEMM:  Cout[m][n] = bias[n] (+ Cinit[m][n])
//                                      + sum_k (2*A[m][k]-1) * Wb[n][k]
// Block = 256 threads = 8 waves; wave -> one 16-row tile, block covers
// 128 rows x 64 cols; K loop in steps of 32 (v_wmma_f32_16x16x32_bf16).
// Used for GEMM1 (logit) and GEMM2 (final out, in place over logit).
// ---------------------------------------------------------------------------
__global__ void __launch_bounds__(256)
gemm_bf16_2xm1(const float* __restrict__ A, const bf16_t* __restrict__ Wb,
               const float* __restrict__ bias, const float* __restrict__ Cinit,
               int useCinit, float* __restrict__ Cout) {
  const int lane = threadIdx.x & 31;
  const int wave = threadIdx.x >> 5;
  const int m0 = (blockIdx.x * 8 + wave) * 16;
  const int n0 = blockIdx.y * 64;
  const int hi  = (lane >= 16) ? 1 : 0;
  const int N   = lane & 15;
  const int Mhi = hi * 8;

  v8f acc[4];
#pragma unroll
  for (int nt = 0; nt < 4; ++nt) {
#pragma unroll
    for (int r = 0; r < 8; ++r) {
      const int row = m0 + r + Mhi;
      const int col = n0 + nt * 16 + N;
      float v = bias[col];
      if (useCinit) v += Cinit[(size_t)row * DIM + col];
      acc[nt][r] = v;
    }
  }

  const float* arow = A + (size_t)(m0 + N) * DIM;
  for (int k0 = 0; k0 < DIM; k0 += 32) {
    v16bf afrag = load_a_2xm1(arow, k0 + hi * 16);
#pragma unroll
    for (int nt = 0; nt < 4; ++nt) {
      const bf16_t* bp = Wb + (size_t)(n0 + nt * 16 + N) * DIM + k0 + hi * 16;
      v16bf bfrag = *(const v16bf*)bp;
      acc[nt] = __builtin_amdgcn_wmma_f32_16x16x32_bf16(
          false, afrag, false, bfrag, (short)0, acc[nt], false, false);
    }
  }

#pragma unroll
  for (int nt = 0; nt < 4; ++nt)
#pragma unroll
    for (int r = 0; r < 8; ++r)
      Cout[(size_t)(m0 + r + Mhi) * DIM + n0 + nt * 16 + N] = acc[nt][r];
}

// ---------------------------------------------------------------------------
// Kernel C: autoregressive Bernoulli sampling, 16 rows per workgroup.
// Prefix accumulators acc[b][i] = logit[b][i] + sum_{j in finished blocks}
// W1[i][j]*buf[b][j] live in WMMA accumulator registers: 64 16x16 tiles,
// tile nt owned by wave (nt&7) slot (nt>>3).  Per 32-column block:
//   1) owning waves stage the block's acc columns to LDS
//   2) wave 0 runs the 32 serial sampling steps with a per-row bitmask
//      (intra-block dot = masked sum over an LDS-resident 32x32 W1 tile;
//       sigmoid(a) > u  <=>  a > log(u/(1-u)))
//   3) all waves rank-32 WMMA-update their future tiles:
//      acc = wmma(samples_as_bf16(0/1), W1b_tile, acc)
// ---------------------------------------------------------------------------
__global__ void __launch_bounds__(256)
ar_sample(const float* __restrict__ logit, const float* __restrict__ W1f,
          const bf16_t* __restrict__ W1b, const float* __restrict__ U,
          float* __restrict__ bufOut) {
  __shared__ float    stage[16][32];
  __shared__ float    w1t[32][32];
  __shared__ float    thr[16][32];
  __shared__ unsigned smask[16];

  const int tid  = threadIdx.x;
  const int lane = tid & 31;
  const int wave = tid >> 5;
  const int row_base = blockIdx.x * 16;
  const int hi  = (lane >= 16) ? 1 : 0;
  const int N   = lane & 15;
  const int Mhi = hi * 8;

  // init accumulators with logit
  v8f accT[8];
#pragma unroll
  for (int tt = 0; tt < 8; ++tt) {
    const int nt = wave + 8 * tt;
#pragma unroll
    for (int r = 0; r < 8; ++r)
      accT[tt][r] = logit[(size_t)(row_base + r + Mhi) * DIM + nt * 16 + N];
  }

  for (int t = 0; t < 32; ++t) {
    const int j0 = 32 * t;

    // (1) stage this block's acc columns (tiles 2t, 2t+1) to LDS
    {
      const int nt0 = 2 * t, nt1 = 2 * t + 1;
      if (wave == (nt0 & 7)) {
        const int tt = nt0 >> 3;
#pragma unroll
        for (int r = 0; r < 8; ++r) stage[r + Mhi][N] = accT[tt][r];
      }
      if (wave == (nt1 & 7)) {
        const int tt = nt1 >> 3;
#pragma unroll
        for (int r = 0; r < 8; ++r) stage[r + Mhi][16 + N] = accT[tt][r];
      }
    }
    // preload intra-block W1 tile (fp32) and thresholds log(u/(1-u))
    for (int idx = tid; idx < 1024; idx += 256) {
      const int ii = idx >> 5, j = idx & 31;
      w1t[ii][j] = W1f[(size_t)(j0 + ii) * DIM + j0 + j];
    }
    for (int idx = tid; idx < 512; idx += 256) {
      const int b = idx >> 5, jj = idx & 31;
      const float uu = U[(size_t)(row_base + b) * DIM + j0 + jj];
      thr[b][jj] = __logf(uu) - __logf(1.0f - uu);
    }
    __syncthreads();

    // (2) serial sampling: wave 0, lane b < 16 handles row b
    if (wave == 0 && lane < 16) {
      unsigned mask = 0;
      for (int ii = 0; ii < 32; ++ii) {
        float a = stage[lane][ii];
        for (int j = 0; j < ii; ++j)
          if (mask & (1u << j)) a += w1t[ii][j];
        if (a > thr[lane][ii]) mask |= (1u << ii);
      }
      smask[lane] = mask;
      float* brow = bufOut + (size_t)(row_base + lane) * DIM + j0;
#pragma unroll
      for (int ii = 0; ii < 32; ii += 4) {
        float4 v;
        v.x = (float)((mask >> (ii + 0)) & 1u);
        v.y = (float)((mask >> (ii + 1)) & 1u);
        v.z = (float)((mask >> (ii + 2)) & 1u);
        v.w = (float)((mask >> (ii + 3)) & 1u);
        *(float4*)(brow + ii) = v;
      }
    }
    __syncthreads();

    // (3) rank-32 WMMA update of all future tiles (uniform per wave)
    const unsigned m = smask[N];
    v16bf afrag;
#pragma unroll
    for (int e = 0; e < 16; ++e) {
      const int K = hi * 16 + e;
      afrag[e] = (__bf16)(float)((m >> K) & 1u);
    }
#pragma unroll
    for (int tt = 0; tt < 8; ++tt) {
      const int nt = wave + 8 * tt;
      if (nt >= 2 * t + 2) {   // wave-uniform condition, EXEC stays full
        const bf16_t* bp = W1b + (size_t)(nt * 16 + N) * DIM + j0 + hi * 16;
        v16bf bfrag = *(const v16bf*)bp;
        accT[tt] = __builtin_amdgcn_wmma_f32_16x16x32_bf16(
            false, afrag, false, bfrag, (short)0, accT[tt], false, false);
      }
    }
    __syncthreads();
  }
}

// ---------------------------------------------------------------------------
extern "C" void kernel_launch(void* const* d_in, const int* in_sizes, int n_in,
                              void* d_out, int out_size, void* d_ws, size_t ws_size,
                              hipStream_t stream) {
  (void)in_sizes; (void)n_in; (void)out_size; (void)ws_size;

  const float* x  = (const float*)d_in[0];   // [B, IN]
  const float* W0 = (const float*)d_in[1];   // [D, IN]
  const float* b0 = (const float*)d_in[2];   // [D]
  const float* W1 = (const float*)d_in[3];   // [D, D]
  const float* b1 = (const float*)d_in[4];   // [D]
  const float* u  = (const float*)d_in[5];   // [B, D]

  float* out = (float*)d_out;                    // first:  out  [B, D]
  float* buf = out + (size_t)BATCH * DIM;        // second: buf  [B, D]

  bf16_t* W0b = (bf16_t*)d_ws;                   // 2 MB
  bf16_t* W1b = W0b + (size_t)DIM * DIM;         // 2 MB

  // P: fp32 -> bf16 weight conversion (row-major = direct B fragments)
  prep_bf16<<<(DIM * DIM) / 256, 256, 0, stream>>>(W0, W1, W0b, W1b);

  // GEMM1: logit = (2x-1) @ W0^T + b0  -> stored in d_out "out" slot
  dim3 ggrid(BATCH / 128, DIM / 64);
  gemm_bf16_2xm1<<<ggrid, 256, 0, stream>>>(x, W0b, b0, out, 0, out);

  // AR sampling: reads logit, writes binary buf
  ar_sample<<<BATCH / 16, 256, 0, stream>>>(out, W1, W1b, u, buf);

  // GEMM2: out = logit + (2*buf-1) @ W1^T + b1  (in place over logit)
  gemm_bf16_2xm1<<<ggrid, 256, 0, stream>>>(buf, W1b, b1, out, 1, out);
}
